// SampleModel_67216238182736
// MI455X (gfx1250) — matrix operand order repaired and checked by reading, not compiled, not run on hardware
//
#include <hip/hip_runtime.h>
#include <hip/hip_bf16.h>
#include <math.h>

// ---------------------------------------------------------------------------
// Problem constants (from reference): features [50000,768] f32,
// centroids [4096,768] f32, TEMPERATURE=0.07, BALANCE=1.0, scalar f32 out.
// ---------------------------------------------------------------------------
#define N_FEAT   50000
#define DIM      768
#define K_CENT   4096
#define DT_CHUNKS (DIM / 32)    // 24 K-chunks of 32 for 16x16x32 WMMA
#define NTILES    (K_CENT / 16) // 256 column tiles
#define INV_T    (1.0f / 0.07f)

typedef __attribute__((ext_vector_type(16))) __bf16 v16bf;
typedef __attribute__((ext_vector_type(8)))  float  v8f;

// Split an f32 into bf16 hi + bf16 lo (hi = rne(x), lo = rne(x - hi)).
__device__ __forceinline__ void splitbf(float f, __bf16& h, __bf16& l) {
  h = (__bf16)f;
  l = (__bf16)(f - (float)h);
}

__device__ __forceinline__ v8f wmma_bf16(v16bf a, v16bf b, v8f c) {
  return __builtin_amdgcn_wmma_f32_16x16x32_bf16(
      /*neg_a=*/false, a, /*neg_b=*/false, b,
      /*c_mod=*/(short)0, c, /*reuse_a=*/false, /*reuse_b=*/false);
}

// ---------------------------------------------------------------------------
// Kernel 1: normalize centroid rows, split to bf16 hi/lo, pre-pack into the
// exact per-lane WMMA operand layouts (ISA 7.12.2) so GEMM waves can load a
// fragment as 32 contiguous bytes per lane.
//   A layout (16x32, 16-bit): lane L (M = L&15); L<16 holds K={0..7,16..23},
//                             L>=16 holds K={8..15,24..31}.
//   B layout (32x16, 16-bit): lane L (N = L&15); L<16 holds K={0..7,16..23},
//                             L>=16 holds K={8..15,24..31}.
// Packed tile index: ((n/16)*DT_CHUNKS + k/32)*32 lanes * 16 bf16.
// ---------------------------------------------------------------------------
__global__ __launch_bounds__(256) void normalize_pack(
    const float* __restrict__ cent,
    __bf16* __restrict__ Bhi, __bf16* __restrict__ Blo,
    __bf16* __restrict__ Ahi, __bf16* __restrict__ Alo) {
  __shared__ float red[256];
  const int n   = blockIdx.x;
  const int tid = threadIdx.x;
  const float* row = cent + (size_t)n * DIM;
  float v0 = row[tid], v1 = row[tid + 256], v2 = row[tid + 512];
  red[tid] = v0 * v0 + v1 * v1 + v2 * v2;
  __syncthreads();
  for (int s = 128; s > 0; s >>= 1) {
    if (tid < s) red[tid] += red[tid + s];
    __syncthreads();
  }
  const float scale = 1.0f / fmaxf(sqrtf(red[0]), 1e-12f);

  const int   ks[3] = {tid, tid + 256, tid + 512};
  const float vs[3] = {v0, v1, v2};
#pragma unroll
  for (int e = 0; e < 3; ++e) {
    const int k = ks[e];
    __bf16 h, l;
    splitbf(vs[e] * scale, h, l);
    const int dt = k >> 5, kl = k & 31;
    const size_t tileBase = (size_t)((n >> 4) * DT_CHUNKS + dt) * 32;
    // B layout
    const int laneB = (n & 15) | (((kl >> 3) & 1) << 4);
    const size_t ib = (tileBase + laneB) * 16 + ((kl & 7) + 8 * ((kl >> 4) & 1));
    Bhi[ib] = h; Blo[ib] = l;
    // A layout
    const int half  = (kl >> 3) & 1;
    const int tA    = (kl & 7) + 8 * ((kl >> 4) & 1);
    const int laneA = (n & 15) + 16 * half;
    const size_t ia = (tileBase + laneA) * 16 + tA;
    Ahi[ia] = h; Alo[ia] = l;
  }
}

// ---------------------------------------------------------------------------
// Kernel 2: fused GEMM + row max/argmax.
// Block = 8 waves * 16 rows = 128 feature rows. Each wave register-blocks
// 8 column tiles (128 centroids) and loops over all 32 n-blocks, keeping a
// running (max, argmax) per row across all 4096 centroids. Features are
// gathered as f32 straight from global (L2-resident) and split on the fly.
// ---------------------------------------------------------------------------
__global__ __launch_bounds__(256) void gemm_rowmax(
    const float* __restrict__ F,
    const __bf16* __restrict__ Bhi, const __bf16* __restrict__ Blo,
    float* __restrict__ mOut, int* __restrict__ iOut) {
  const int tid  = threadIdx.x;
  const int wv   = tid >> 5;
  const int lane = tid & 31;
  const int mrow = lane & 15;   // A-layout row this lane feeds
  const int kSel = lane >> 4;   // which K interleave half
  const int rowBase = blockIdx.x * 128 + wv * 16;
  int row = rowBase + mrow;
  if (row >= N_FEAT) row = N_FEAT - 1;  // clamp padded rows (stores guarded)
  const float* fr = F + (size_t)row * DIM;

  const v16bf* BH = (const v16bf*)Bhi;
  const v16bf* BL = (const v16bf*)Blo;

  float runv[8];
  int   runi[8];
#pragma unroll
  for (int v = 0; v < 8; ++v) { runv[v] = -3.4e38f; runi[v] = 0; }

  for (int nb = 0; nb < 32; ++nb) {
    v8f C[8];
#pragma unroll
    for (int j = 0; j < 8; ++j) { v8f z = {}; C[j] = z; }

    for (int dt = 0; dt < DT_CHUNKS; ++dt) {
      const int kb = dt * 32;
      // Gather this lane's 16 A elements (two 32B runs) and split hi/lo.
      const float4* p0 = (const float4*)(fr + kb + 8 * kSel);
      const float4* p1 = (const float4*)(fr + kb + 16 + 8 * kSel);
      const float4 x0 = p0[0], x1 = p0[1], y0 = p1[0], y1 = p1[1];
      const float xs[16] = {x0.x, x0.y, x0.z, x0.w, x1.x, x1.y, x1.z, x1.w,
                            y0.x, y0.y, y0.z, y0.w, y1.x, y1.y, y1.z, y1.w};
      v16bf ah, al;
#pragma unroll
      for (int t = 0; t < 16; ++t) {
        __bf16 h, l;
        splitbf(xs[t], h, l);
        ah[t] = h; al[t] = l;
      }
      if (dt + 1 < DT_CHUNKS)  // hint the next B stream (global_prefetch_b8)
        __builtin_prefetch(
            (const void*)(BH + ((size_t)(nb * 8 * DT_CHUNKS + dt + 1) * 32 + lane)), 0, 1);

#pragma unroll
      for (int j = 0; j < 8; ++j) {
        const int nt = nb * 8 + j;
        const size_t bidx = (size_t)(nt * DT_CHUNKS + dt) * 32 + lane;
        const v16bf bh = BH[bidx];
        const v16bf bl = BL[bidx];
        C[j] = wmma_bf16(ah, bh, C[j]);   // hi*hi
        C[j] = wmma_bf16(ah, bl, C[j]);   // hi*lo
        C[j] = wmma_bf16(al, bh, C[j]);   // lo*hi  (~f32-accurate dot)
      }
    }

    // Epilogue: reduce each 16x16 C tile to per-row (max, argmax).
    // C layout: VGPR v, lane L -> element (M = v + 8*(L>>4), N = L&15).
#pragma unroll
    for (int j = 0; j < 8; ++j) {
      const int colBase = nb * 128 + j * 16 + (lane & 15);
#pragma unroll
      for (int v = 0; v < 8; ++v) {
        float val = C[j][v] * INV_T;
        int   col = colBase;
#pragma unroll
        for (int s = 1; s < 16; s <<= 1) {  // stays within 16-lane half
          const float ov = __shfl_xor(val, s, 32);
          const int   oc = __shfl_xor(col, s, 32);
          if (ov > val || (ov == val && oc < col)) { val = ov; col = oc; }
        }
        if (val > runv[v] || (val == runv[v] && col < runi[v])) {
          runv[v] = val; runi[v] = col;
        }
      }
    }
  }

  // Lane 0 holds rows M=0..7, lane 16 holds rows M=8..15.
  if ((lane & 15) == 0) {
    const int half = lane >> 4;
#pragma unroll
    for (int v = 0; v < 8; ++v) {
      const int r = rowBase + half * 8 + v;
      if (r < N_FEAT) { mOut[r] = runv[v]; iOut[r] = runi[v]; }
    }
  }
}

// ---------------------------------------------------------------------------
// Kernel 3: S[k] = sum_j exp(dot(c_j, c_k)/T). One block per 16-column tile;
// 8 waves stripe the 4096 j-rows; same split-bf16 WMMA core with an
// exp/column-sum epilogue. Wave partials combined through LDS.
// ---------------------------------------------------------------------------
__global__ __launch_bounds__(256) void cent_expsum(
    const __bf16* __restrict__ Ahi, const __bf16* __restrict__ Alo,
    const __bf16* __restrict__ Bhi, const __bf16* __restrict__ Blo,
    float* __restrict__ S) {
  __shared__ float red[8 * 16];
  const int tid  = threadIdx.x;
  const int wv   = tid >> 5;
  const int lane = tid & 31;
  const int nt   = blockIdx.x;  // column tile 0..255
  const v16bf* AH = (const v16bf*)Ahi;
  const v16bf* AL = (const v16bf*)Alo;
  const v16bf* BH = (const v16bf*)Bhi;
  const v16bf* BL = (const v16bf*)Blo;

  float colsum = 0.0f;
  for (int pass = 0; pass < 32; ++pass) {
    const int jt = pass * 8 + wv;  // row tile 0..255
    v8f C = {};
    for (int dt = 0; dt < DT_CHUNKS; ++dt) {
      const size_t aidx = (size_t)(jt * DT_CHUNKS + dt) * 32 + lane;
      const size_t bidx = (size_t)(nt * DT_CHUNKS + dt) * 32 + lane;
      const v16bf ah = AH[aidx], al = AL[aidx];
      const v16bf bh = BH[bidx], bl = BL[bidx];
      C = wmma_bf16(ah, bh, C);
      C = wmma_bf16(ah, bl, C);
      C = wmma_bf16(al, bh, C);
    }
#pragma unroll
    for (int v = 0; v < 8; ++v) colsum += __expf(C[v] * INV_T);
  }
  colsum += __shfl_xor(colsum, 16, 32);  // combine the two row halves
  if (lane < 16) red[wv * 16 + lane] = colsum;
  __syncthreads();
  if (tid < 16) {
    float s = 0.0f;
#pragma unroll
    for (int w = 0; w < 8; ++w) s += red[w * 16 + tid];
    S[nt * 16 + tid] = s;
  }
}

// ---------------------------------------------------------------------------
// Kernels 4/5: deterministic two-stage loss reduction.
// J_i = log(exp(m_i)) - log(exp(m_i) + S[k_i]);  out = -mean(J).
// ---------------------------------------------------------------------------
__global__ __launch_bounds__(256) void loss_partial(
    const float* __restrict__ m, const int* __restrict__ idx,
    const float* __restrict__ S, float* __restrict__ part) {
  __shared__ float red[256];
  const int tid = threadIdx.x;
  float acc = 0.0f;
  for (int i = blockIdx.x * 256 + tid; i < N_FEAT; i += gridDim.x * 256) {
    const float p = __expf(m[i]);
    const float s = S[idx[i]];             // BALANCE == 1.0
    acc += __logf(p) - __logf(p + s);
  }
  red[tid] = acc;
  __syncthreads();
  for (int s2 = 128; s2 > 0; s2 >>= 1) {
    if (tid < s2) red[tid] += red[tid + s2];
    __syncthreads();
  }
  if (tid == 0) part[blockIdx.x] = red[0];
}

__global__ __launch_bounds__(256) void loss_final(
    const float* __restrict__ part, int nPart, float* __restrict__ out) {
  __shared__ float red[256];
  const int tid = threadIdx.x;
  float a = 0.0f;
  for (int i = tid; i < nPart; i += 256) a += part[i];
  red[tid] = a;
  __syncthreads();
  for (int s = 128; s > 0; s >>= 1) {
    if (tid < s) red[tid] += red[tid + s];
    __syncthreads();
  }
  if (tid == 0) out[0] = -(red[0] / (float)N_FEAT);
}

// ---------------------------------------------------------------------------
extern "C" void kernel_launch(void* const* d_in, const int* in_sizes, int n_in,
                              void* d_out, int out_size, void* d_ws, size_t ws_size,
                              hipStream_t stream) {
  (void)in_sizes; (void)n_in; (void)out_size; (void)ws_size;
  const float* features  = (const float*)d_in[0];  // [50000, 768]
  const float* centroids = (const float*)d_in[1];  // [4096, 768]

  char* ws = (char*)d_ws;
  size_t off = 0;
  auto alloc = [&](size_t bytes) -> char* {
    char* p = ws + off;
    off += (bytes + 255) & ~(size_t)255;
    return p;
  };
  __bf16* Bhi = (__bf16*)alloc((size_t)K_CENT * DIM * 2);  // 6.3 MB
  __bf16* Blo = (__bf16*)alloc((size_t)K_CENT * DIM * 2);
  __bf16* Ahi = (__bf16*)alloc((size_t)K_CENT * DIM * 2);
  __bf16* Alo = (__bf16*)alloc((size_t)K_CENT * DIM * 2);
  float*  S   = (float*)alloc((size_t)K_CENT * 4);
  float*  mA  = (float*)alloc((size_t)N_FEAT * 4);
  int*    iA  = (int*)alloc((size_t)N_FEAT * 4);
  float*  prt = (float*)alloc(256 * 4);

  normalize_pack<<<K_CENT, 256, 0, stream>>>(centroids, Bhi, Blo, Ahi, Alo);
  gemm_rowmax<<<(N_FEAT + 127) / 128, 256, 0, stream>>>(features, Bhi, Blo, mA, iA);
  cent_expsum<<<NTILES, 256, 0, stream>>>(Ahi, Alo, Bhi, Blo, S);
  const int nPart = 128;
  loss_partial<<<nPart, 256, 0, stream>>>(mA, iA, S, prt);
  loss_final<<<1, 256, 0, stream>>>(prt, nPart, (float*)d_out);
}
